// FastMultiHeadAttention_21036749815890
// MI455X (gfx1250) — matrix operand order repaired and checked by reading, not compiled, not run on hardware
//
#include <hip/hip_runtime.h>
#include <hip/hip_bf16.h>
#include <stdint.h>

typedef unsigned short u16;
typedef __attribute__((ext_vector_type(16))) __bf16 v16bf;
typedef __attribute__((ext_vector_type(8)))  __bf16 v8bf;
typedef __attribute__((ext_vector_type(8)))  float  v8f;
typedef __attribute__((ext_vector_type(4)))  unsigned int v4u;
typedef __attribute__((ext_vector_type(8)))  int v8i;
typedef __attribute__((ext_vector_type(4)))  int v4i;

#define T_SEQ 4096
#define C_DIM 768
#define C3    2304
#define HEADS 12
#define HD    64
#define NEGV  (-10000.0f)

// ---------- helpers ----------

__device__ __forceinline__ u16 f2bf(float x) {
  union { float f; uint32_t u; } c; c.f = x;
  uint32_t r = c.u + 0x7FFFu + ((c.u >> 16) & 1u);   // round-nearest-even
  return (u16)(r >> 16);
}

// A-fragment pattern: 16 contiguous bf16 at p (two 16B aligned chunks)
__device__ __forceinline__ v16bf frag_pair(const u16* p) {
  v8bf a = *(const v8bf*)(p);
  v8bf b = *(const v8bf*)(p + 8);
  return __builtin_shufflevector(a, b, 0,1,2,3,4,5,6,7,8,9,10,11,12,13,14,15);
}

// B-fragment pattern: 16 contiguous bf16 at p
__device__ __forceinline__ v16bf frag_contig(const u16* p) {
  v8bf a = *(const v8bf*)(p);
  v8bf b = *(const v8bf*)(p + 8);
  return __builtin_shufflevector(a, b, 0,1,2,3,4,5,6,7,8,9,10,11,12,13,14,15);
}

__device__ __forceinline__ v8f wmma_bf16(v16bf a, v16bf b, v8f c) {
  return __builtin_amdgcn_wmma_f32_16x16x32_bf16(false, a, false, b, (short)0, c,
                                                 false, false);
}

// Issue a TDM 2D tile load: int32 tile [16 rows x 32 cols] of the attention
// mask at (row0, col0) into LDS at byte offset lds_off. Wave-uniform D#.
__device__ __forceinline__ void tdm_load_mask_tile(const int* gbase, unsigned lds_off,
                                                   int row0, int col0) {
  unsigned long long ga =
      (unsigned long long)(uintptr_t)(gbase + ((size_t)row0 << 12) + col0);
  v4u g0;
  g0[0] = 1u;                                        // count=1 (valid user D#)
  g0[1] = lds_off;                                   // lds_addr (bytes)
  g0[2] = (unsigned)(ga & 0xFFFFFFFFu);              // global_addr[31:0]
  g0[3] = (unsigned)((ga >> 32) & 0x01FFFFFFu)       // global_addr[56:32]
          | (2u << 30);                              // type=2 ("image")
  v8i g1;
  g1[0] = 0x00020000;                                // data_size=2 -> 4B elems
  g1[1] = (int)((T_SEQ & 0xFFFFu) << 16);            // tensor_dim0 lo16
  g1[2] = (int)((T_SEQ >> 16) | ((T_SEQ & 0xFFFFu) << 16)); // dim0 hi | dim1 lo
  g1[3] = (int)((T_SEQ >> 16) | (32u << 16));        // dim1 hi | tile_dim0=32
  g1[4] = 16;                                        // tile_dim1=16, tile_dim2=0
  g1[5] = T_SEQ;                                     // tensor_dim0_stride lo32
  g1[6] = 0;                                         // stride hi | dim1_stride lo
  g1[7] = 0;
  v4i z4 = {0, 0, 0, 0};
#if __clang_major__ >= 23
  v8i z8 = {0, 0, 0, 0, 0, 0, 0, 0};
  __builtin_amdgcn_tensor_load_to_lds(g0, g1, z4, z4, z8, 0);
#else
  __builtin_amdgcn_tensor_load_to_lds(g0, g1, z4, z4, 0);
#endif
}

// ---------- elementwise prep kernels ----------

__global__ void build_x_kernel(const float* __restrict__ q, const float* __restrict__ k,
                               const float* __restrict__ v, u16* __restrict__ X) {
  int idx = blockIdx.x * 256 + threadIdx.x;        // exactly T_SEQ*C3 threads
  int t = idx / C3;
  int c = idx - t * C3;
  float val;
  if (c < C_DIM)            val = q[(size_t)t * C_DIM + c];
  else if (c < 2 * C_DIM)   val = k[(size_t)t * C_DIM + (c - C_DIM)];
  else                      val = v[(size_t)t * C_DIM + (c - 2 * C_DIM)];
  X[idx] = f2bf(val);
}

__global__ void convert_bf16_kernel(const float* __restrict__ src, u16* __restrict__ dst, int n) {
  int idx = blockIdx.x * 256 + threadIdx.x;
  if (idx < n) dst[idx] = f2bf(src[idx]);
}

// Vt[h][d][s] = QKV[s][h*192 + 128 + d]   (bf16 -> bf16 gather-transpose)
__global__ void build_vt_kernel(const u16* __restrict__ QKV, u16* __restrict__ Vt) {
  int idx = blockIdx.x * 256 + threadIdx.x;        // exactly HEADS*HD*T_SEQ threads
  int s  = idx & (T_SEQ - 1);
  int hd = idx >> 12;
  int d  = hd & (HD - 1);
  int h  = hd >> 6;
  Vt[idx] = QKV[(size_t)s * C3 + h * 192 + 128 + d];
}

// ---------- WMMA GEMM:  C[m][n] = sum_k A[m][k] * W[n][k] ----------
// block = 256 threads = 8 waves; block tile 64(M) x 256(N); wave tile 32 x 64.
// Each B-fragment is reused by two A-fragments (8 WMMAs per k-step per wave).

template <bool OUT_BF16>
__global__ __launch_bounds__(256)
void gemm_wmma_kernel(const u16* __restrict__ A, const u16* __restrict__ W,
                      void* __restrict__ Cout, int M, int N, int K) {
  const int tid  = threadIdx.x;
  const int lane = tid & 31;
  const int wave = tid >> 5;
  const int l16  = lane & 15;
  const int hi   = lane >> 4;
  const int wm   = wave & 1;            // 2 M sub-tiles of 32
  const int wn   = wave >> 1;           // 4 N sub-tiles of 64
  const int m0   = blockIdx.x * 64  + wm * 32;
  const int n0   = blockIdx.y * 256 + wn * 64;

  v8f acc[2][4] = {};

  const u16* Arow0 = A + (size_t)(m0 +  0 + l16) * K + hi * 16;
  const u16* Arow1 = A + (size_t)(m0 + 16 + l16) * K + hi * 16;
  const u16* B0 = W + (size_t)(n0 +  0 + l16) * K + hi * 16;
  const u16* B1 = W + (size_t)(n0 + 16 + l16) * K + hi * 16;
  const u16* B2 = W + (size_t)(n0 + 32 + l16) * K + hi * 16;
  const u16* B3 = W + (size_t)(n0 + 48 + l16) * K + hi * 16;

  for (int k0 = 0; k0 < K; k0 += 32) {
    __builtin_prefetch(Arow0 + k0 + 256, 0, 1);
    __builtin_prefetch(Arow1 + k0 + 256, 0, 1);
    v16bf a0 = frag_pair(Arow0 + k0);
    v16bf a1 = frag_pair(Arow1 + k0);
    v16bf b;
    b = frag_contig(B0 + k0);
    acc[0][0] = wmma_bf16(a0, b, acc[0][0]);
    acc[1][0] = wmma_bf16(a1, b, acc[1][0]);
    b = frag_contig(B1 + k0);
    acc[0][1] = wmma_bf16(a0, b, acc[0][1]);
    acc[1][1] = wmma_bf16(a1, b, acc[1][1]);
    b = frag_contig(B2 + k0);
    acc[0][2] = wmma_bf16(a0, b, acc[0][2]);
    acc[1][2] = wmma_bf16(a1, b, acc[1][2]);
    b = frag_contig(B3 + k0);
    acc[0][3] = wmma_bf16(a0, b, acc[0][3]);
    acc[1][3] = wmma_bf16(a1, b, acc[1][3]);
  }

#pragma unroll
  for (int i = 0; i < 2; ++i) {
#pragma unroll
    for (int j = 0; j < 4; ++j) {
#pragma unroll
      for (int r = 0; r < 8; ++r) {
        int m = m0 + i * 16 + r + 8 * hi;
        int n = n0 + j * 16 + l16;
        float val = acc[i][j][r];
        if (OUT_BF16) ((u16*)Cout)[(size_t)m * N + n]   = f2bf(val);
        else          ((float*)Cout)[(size_t)m * N + n] = val;
      }
    }
  }
}

// ---------- flash attention (per wave: 16 query rows, full hd=64) ----------
// grid = (T/128, HEADS), block = 256 (8 waves).
// Mask tiles streamed with the Tensor Data Mover into per-wave LDS.
// Output written SCRAMBLED per reference: Y.flat[(h*64+d)*T + t] = o[h][t][d].

__global__ __launch_bounds__(256)
void attn_wmma_kernel(const u16* __restrict__ QKV, const u16* __restrict__ Vt,
                      const int* __restrict__ amask, const int* __restrict__ mask_future,
                      u16* __restrict__ Y) {
  const int h    = blockIdx.y;
  const int tid  = threadIdx.x;
  const int lane = tid & 31;
  const int wave = tid >> 5;
  const int l16  = lane & 15;
  const int hi   = lane >> 4;
  const int qbase = blockIdx.x * 128 + wave * 16;
  const bool causal = (mask_future[0] != 0);

  __shared__ u16 pbuf[8][16 * 32];      // per-wave P tile, row-major 16x32 (8 KB)
  __shared__ int mbuf[8][16 * 32];      // per-wave TDM mask tile (16 KB)
  u16* pb = pbuf[wave];
  const int* mb = mbuf[wave];
  const unsigned mb_off = (unsigned)(uintptr_t)&mbuf[wave][0];  // LDS byte offset

  // Q A-fragments for the two K=32 contraction steps over hd=64
  const u16* Qrow = QKV + (size_t)(qbase + l16) * C3 + h * 192;
  v16bf qf0 = frag_pair(Qrow + hi * 16);
  v16bf qf1 = frag_pair(Qrow + 32 + hi * 16);

  float mrow[8], lrow[8];
  v8f oacc[4] = {};
#pragma unroll
  for (int r = 0; r < 8; ++r) { mrow[r] = -3.0e30f; lrow[r] = 0.0f; }

  const int qmax = qbase + 15;
  for (int kb = 0; kb < T_SEQ; kb += 32) {
    if (causal && kb > qmax) break;    // uniform per wave

    // Async TDM load of this wave's 16x32 int32 mask tile into LDS.
    // Previous iteration's LDS reads must be complete before the DMA lands.
    asm volatile("s_wait_dscnt 0x0" ::: "memory");
    tdm_load_mask_tile(amask, mb_off, qbase, kb);

    // ---- S = Q K^T for 32 keys (overlaps with the TDM transfer) ----
    v8f s0 = {}, s1 = {};
    {
      const u16* Kr = QKV + (size_t)(kb + l16) * C3 + h * 192 + 64 + hi * 16;
      s0 = wmma_bf16(qf0, frag_contig(Kr), s0);
      s0 = wmma_bf16(qf1, frag_contig(Kr + 32), s0);
    }
    {
      const u16* Kr = QKV + (size_t)(kb + 16 + l16) * C3 + h * 192 + 64 + hi * 16;
      s1 = wmma_bf16(qf0, frag_contig(Kr), s1);
      s1 = wmma_bf16(qf1, frag_contig(Kr + 32), s1);
    }

    __builtin_amdgcn_s_wait_tensorcnt(0);
    asm volatile("" ::: "memory");

    // ---- scale + causal + mask (mask read from LDS tile) ----
    float e0[8], e1[8];
    const int n0c = kb + l16;
    const int n1c = kb + 16 + l16;
#pragma unroll
    for (int r = 0; r < 8; ++r) {
      int mrel = r + 8 * hi;
      int m = qbase + mrel;
      float x0 = s0[r] * 0.125f;                 // (hd^-1/4)^2 = 1/8
      float x1 = s1[r] * 0.125f;
      if (causal && n0c > m) x0 += NEGV;
      if (causal && n1c > m) x1 += NEGV;
      if (mb[mrel * 32 + l16] == 0)      x0 = NEGV;
      if (mb[mrel * 32 + 16 + l16] == 0) x1 = NEGV;
      e0[r] = x0;
      e1[r] = x1;
    }

    // ---- online softmax over the 32 columns ----
#pragma unroll
    for (int r = 0; r < 8; ++r) {
      float mx = fmaxf(e0[r], e1[r]);
#pragma unroll
      for (int off = 8; off > 0; off >>= 1)
        mx = fmaxf(mx, __shfl_xor(mx, off, 16));   // row spread across 16 lanes
      float mn = fmaxf(mrow[r], mx);
      float al = __expf(mrow[r] - mn);
      float p0 = __expf(e0[r] - mn);
      float p1 = __expf(e1[r] - mn);
      float rs = p0 + p1;
#pragma unroll
      for (int off = 8; off > 0; off >>= 1)
        rs += __shfl_xor(rs, off, 16);
      lrow[r] = lrow[r] * al + rs;
      mrow[r] = mn;
      int mrel = r + 8 * hi;
      pb[mrel * 32 + l16]      = f2bf(p0);
      pb[mrel * 32 + 16 + l16] = f2bf(p1);
#pragma unroll
      for (int j = 0; j < 4; ++j) oacc[j][r] *= al;
    }

    asm volatile("s_wait_dscnt 0x0" ::: "memory");  // per-wave LDS store->load

    // ---- PV: A = P(16x32) from LDS, B = Vt (contiguous in k) ----
    v16bf pf = frag_pair(pb + l16 * 32 + hi * 16);
#pragma unroll
    for (int j = 0; j < 4; ++j) {
      const u16* Vr = Vt + ((size_t)(h * HD + j * 16 + l16)) * T_SEQ + kb + hi * 16;
      oacc[j] = wmma_bf16(pf, frag_contig(Vr), oacc[j]);
    }
  }

  // ---- epilogue: normalize + scrambled store (reference's transpose.reshape) ----
#pragma unroll
  for (int j = 0; j < 4; ++j) {
#pragma unroll
    for (int r = 0; r < 8; ++r) {
      int m = qbase + r + 8 * hi;
      int d = j * 16 + l16;
      float val = oacc[j][r] / lrow[r];
      Y[((size_t)(h * HD + d)) * T_SEQ + m] = f2bf(val);
    }
  }
}

// ---------- host launcher ----------

extern "C" void kernel_launch(void* const* d_in, const int* in_sizes, int n_in,
                              void* d_out, int out_size, void* d_ws, size_t ws_size,
                              hipStream_t stream) {
  (void)in_sizes; (void)n_in; (void)out_size; (void)ws_size;
  const float* q          = (const float*)d_in[0];
  const float* k          = (const float*)d_in[1];
  const float* v          = (const float*)d_in[2];
  const int*   attn_mask  = (const int*)d_in[3];
  const int*   mask_fut   = (const int*)d_in[4];
  const float* W_qkv      = (const float*)d_in[5];
  const float* W_out      = (const float*)d_in[6];

  size_t off = 0;
  char* base = (char*)d_ws;
  auto carve = [&](size_t bytes) -> void* {
    void* p = base + off;
    off += (bytes + 255) & ~(size_t)255;
    return p;
  };
  u16* Xbf    = (u16*)carve((size_t)T_SEQ * C3 * 2);        // 18.9 MB
  u16* Wqkvbf = (u16*)carve((size_t)C3 * C3 * 2);           // 10.6 MB
  u16* Woutbf = (u16*)carve((size_t)C_DIM * C_DIM * 2);     //  1.2 MB
  u16* QKVbf  = (u16*)carve((size_t)T_SEQ * C3 * 2);        // 18.9 MB
  u16* Vtb    = (u16*)carve((size_t)HEADS * HD * T_SEQ * 2);//  6.3 MB
  u16* Ybf    = (u16*)carve((size_t)T_SEQ * C_DIM * 2);     //  6.3 MB

  // 1) concat(q,k,v) -> bf16
  build_x_kernel<<<(T_SEQ * C3) / 256, 256, 0, stream>>>(q, k, v, Xbf);
  // 2) weights -> bf16
  convert_bf16_kernel<<<(C3 * C3 + 255) / 256, 256, 0, stream>>>(W_qkv, Wqkvbf, C3 * C3);
  convert_bf16_kernel<<<(C_DIM * C_DIM + 255) / 256, 256, 0, stream>>>(W_out, Woutbf, C_DIM * C_DIM);
  // 3) QKV = X * Wqkv^T  (M=4096, N=2304, K=2304), bf16 out
  gemm_wmma_kernel<true><<<dim3(T_SEQ / 64, C3 / 256), 256, 0, stream>>>(
      Xbf, Wqkvbf, QKVbf, T_SEQ, C3, C3);
  // 4) per-head V transpose for contiguous PV B-fragments
  build_vt_kernel<<<(HEADS * HD * T_SEQ) / 256, 256, 0, stream>>>(QKVbf, Vtb);
  // 5) flash attention (TDM-streamed mask) -> scrambled Y (bf16)
  attn_wmma_kernel<<<dim3(T_SEQ / 128, HEADS), 256, 0, stream>>>(
      QKVbf, Vtb, attn_mask, mask_fut, Ybf);
  // 6) final = Y * Wout^T  (M=4096, N=768, K=768), f32 out
  gemm_wmma_kernel<false><<<dim3(T_SEQ / 64, C_DIM / 256), 256, 0, stream>>>(
      Ybf, Woutbf, d_out, T_SEQ, C_DIM, C_DIM);
}
